// TransRScore_76124000354696
// MI455X (gfx1250) — matrix operand order repaired and checked by reading, not compiled, not run on hardware
//
#include <hip/hip_runtime.h>
#include <hip/hip_bf16.h>

typedef __attribute__((ext_vector_type(16))) _Float16 v16h;
typedef __attribute__((ext_vector_type(8)))  float    v8f;

#define D_DIM   32
#define GAMMA_V 12.0f

// ---------------- Pass 1: bucket edges by relation ----------------

__global__ void zero_cnt_kernel(int* cnt, int n) {
    int i = blockIdx.x * blockDim.x + threadIdx.x;
    if (i < n) cnt[i] = 0;
}

__global__ void hist_kernel(const int* __restrict__ rel_id, int* cnt, int E) {
    int i = blockIdx.x * blockDim.x + threadIdx.x;
    if (i < E) atomicAdd(&cnt[rel_id[i]], 1);
}

// One block of 1024 threads: exclusive scan over nrel counters.
__global__ void scan_kernel(const int* __restrict__ cnt, int* offs, int* cursor,
                            int nrel, int E) {
    __shared__ int buf[1024];
    int t = threadIdx.x;
    if (nrel <= 1024) {
        int v = (t < nrel) ? cnt[t] : 0;
        buf[t] = v;
        __syncthreads();
        // Hillis-Steele inclusive scan
        for (int d = 1; d < 1024; d <<= 1) {
            int add = (t >= d) ? buf[t - d] : 0;
            __syncthreads();
            buf[t] += add;
            __syncthreads();
        }
        if (t < nrel) {
            int excl = buf[t] - v;
            offs[t]   = excl;
            cursor[t] = excl;
        }
        if (t == 0) offs[nrel] = E;
    } else if (t == 0) { // fallback, never hit for N_REL=1000
        int acc = 0;
        for (int r = 0; r < nrel; ++r) { int c = cnt[r]; offs[r] = acc; cursor[r] = acc; acc += c; }
        offs[nrel] = acc;
    }
}

__global__ void scatter_kernel(const int* __restrict__ rel_id, int* cursor,
                               int* perm, int E) {
    int i = blockIdx.x * blockDim.x + threadIdx.x;
    if (i < E) {
        int p = atomicAdd(&cursor[rel_id[i]], 1);
        perm[p] = i;
    }
}

// ---------------- Pass 2: per-relation GEMM + score via WMMA ----------------
//
// One block per relation, 128 threads = 4 waves. W_r converted to f16 in LDS
// once. Each wave owns 16-edge tiles: builds X (32 x 16) f16 B-fragment from
// node_emb[src]-node_emb[dst] gathers, runs two v_wmma_f32_16x16x32_f16
// (M rows 0-15 and 16-31, K=32 == D_ENT in a single WMMA), then reduces
// |y + r| across the D-fragment layout and writes gamma - L1.

__global__ __launch_bounds__(128)
void transr_wmma_kernel(const float* __restrict__ node_emb,
                        const float* __restrict__ rel_emb,
                        const float* __restrict__ transfer_w,
                        const int*   __restrict__ src,
                        const int*   __restrict__ dst,
                        const int*   __restrict__ perm,
                        const int*   __restrict__ offs,
                        float*       __restrict__ out) {
    __shared__ _Float16 Wh[D_DIM * D_DIM];  // 2 KB: W_r in f16, row-major [r][d]
    __shared__ float    relrow[D_DIM];

    const int rel = blockIdx.x;
    const int t   = threadIdx.x;

    const float* W = transfer_w + (size_t)rel * (D_DIM * D_DIM);
    for (int i = t; i < D_DIM * D_DIM; i += 128) Wh[i] = (_Float16)W[i];
    if (t < D_DIM) relrow[t] = rel_emb[rel * D_DIM + t];
    __syncthreads();

    const int start = offs[rel];
    const int cnt   = offs[rel + 1] - start;
    if (cnt <= 0) return;

    const int lane   = t & 31;
    const int wave   = t >> 5;
    const int nwaves = 4;

    // A fragments (16x32 f16, ISA layout): lane L<16 -> row M=L, K in
    // {0..7,16..23}; lane L+16 -> row M=L, K in {8..15,24..31}.
    const int mA = lane & 15;
    const int kb = (lane < 16) ? 0 : 8;
    v16h alo, ahi;
#pragma unroll
    for (int i = 0; i < 8; ++i) {
        alo[i]     = Wh[mA * 32 + kb + i];
        alo[8 + i] = Wh[mA * 32 + kb + 16 + i];
        ahi[i]     = Wh[(mA + 16) * 32 + kb + i];
        ahi[8 + i] = Wh[(mA + 16) * 32 + kb + 16 + i];
    }

    // B fragment (32x16 f16): lane L<16 -> column N=L, K in {0..7,16..23};
    // lane L+16 -> column N=L, K in {8..15,24..31} (same K order as A).
    const int kb2    = (lane < 16) ? 0 : 8;
    const int ntiles = (cnt + 15) >> 4;

    for (int tile = wave; tile < ntiles; tile += nwaves) {
        const int col  = tile * 16 + (lane & 15);
        int       eidx = -1;
        v16h b = {};
        if (col < cnt) {
            eidx = perm[start + col];
            const int s = src[eidx];
            const int d = dst[eidx];
            const float* hp = node_emb + (size_t)s * D_DIM + kb2;
            const float* tp = node_emb + (size_t)d * D_DIM + kb2;
#pragma unroll
            for (int i = 0; i < 8; ++i) {
                b[i]     = (_Float16)(hp[i] - tp[i]);
                b[8 + i] = (_Float16)(hp[16 + i] - tp[16 + i]);
            }
        }

        v8f c0 = {};
        v8f c1 = {};
        // 8 args: (neg_a, A, neg_b, B, c_mod, C, reuse_a, reuse_b)
        c0 = __builtin_amdgcn_wmma_f32_16x16x32_f16(false, alo, false, b,
                                                    (short)0, c0, false, false);
        c1 = __builtin_amdgcn_wmma_f32_16x16x32_f16(false, ahi, false, b,
                                                    (short)0, c1, false, false);

        // D layout: VGPR v -> M=v (lanes 0-15) / M=v+8 (lanes 16-31), N = lane%16.
        const int ro  = (lane < 16) ? 0 : 8;
        float     sum = 0.0f;
#pragma unroll
        for (int v = 0; v < 8; ++v) {
            sum += fabsf(c0[v] + relrow[ro + v]);        // rows 0..15
            sum += fabsf(c1[v] + relrow[16 + ro + v]);   // rows 16..31
        }
        sum += __shfl_xor(sum, 16, 32);  // combine the two half-wave partials

        if (eidx >= 0 && lane < 16) out[eidx] = GAMMA_V - sum;
    }
}

// ---------------- host-side launch ----------------

extern "C" void kernel_launch(void* const* d_in, const int* in_sizes, int n_in,
                              void* d_out, int out_size, void* d_ws, size_t ws_size,
                              hipStream_t stream) {
    const float* node_emb   = (const float*)d_in[0];
    const float* rel_emb    = (const float*)d_in[1];
    const float* transfer_w = (const float*)d_in[2];
    const int*   src        = (const int*)d_in[3];
    const int*   dst        = (const int*)d_in[4];
    const int*   rel_id     = (const int*)d_in[5];
    float*       out        = (float*)d_out;

    const int E    = in_sizes[3];
    const int nrel = in_sizes[1] / D_DIM;

    // workspace layout (ints): cnt[nrel] | offs[nrel+1] | cursor[nrel] | perm[E]
    int* cnt    = (int*)d_ws;
    int* offs   = cnt + nrel;
    int* cursor = offs + (nrel + 1);
    int* perm   = cursor + nrel;

    zero_cnt_kernel<<<(nrel + 255) / 256, 256, 0, stream>>>(cnt, nrel);
    hist_kernel<<<(E + 255) / 256, 256, 0, stream>>>(rel_id, cnt, E);
    scan_kernel<<<1, 1024, 0, stream>>>(cnt, offs, cursor, nrel, E);
    scatter_kernel<<<(E + 255) / 256, 256, 0, stream>>>(rel_id, cursor, perm, E);
    transr_wmma_kernel<<<nrel, 128, 0, stream>>>(node_emb, rel_emb, transfer_w,
                                                 src, dst, perm, offs, out);
}
